// EWGNN_84490596647563
// MI455X (gfx1250) — matrix operand order repaired and checked by reading, not compiled, not run on hardware
//
#include <hip/hip_runtime.h>

// ---------------- problem constants ----------------
#define NN    50000
#define EE    600000
#define INF_  256
#define HID   128
#define HEADS 2
#define QKV   256   // HEADS*HID
#define OUTF  64
#define LN_EPS 1e-5f

typedef __attribute__((ext_vector_type(16))) __bf16 v16bf;
typedef __attribute__((ext_vector_type(8)))  __bf16 v8bf;
typedef __attribute__((ext_vector_type(8)))  float  v8f;
typedef int v4i_vs __attribute__((vector_size(16)));  // matches builtin param

#ifndef __has_builtin
#define __has_builtin(x) 0
#endif
#if __has_builtin(__builtin_amdgcn_global_load_async_to_lds_b128) && \
    __has_builtin(__builtin_amdgcn_s_wait_asynccnt)
#define USE_ASYNC_LDS 1
#else
#define USE_ASYNC_LDS 0
#endif

#if USE_ASYNC_LDS
#define GASYNC_B128(gp, lp)                                  \
  __builtin_amdgcn_global_load_async_to_lds_b128(            \
      (__attribute__((address_space(1))) v4i_vs*)(gp),       \
      (__attribute__((address_space(3))) v4i_vs*)(lp), 0, 0)
#endif

// ---------------- bf16 split helpers (RNE via bit math) ----------------------
__device__ __forceinline__ __bf16 to_bf16(float f) {
  unsigned u = __float_as_uint(f);
  u += 0x7FFFu + ((u >> 16) & 1u);
  unsigned short s = (unsigned short)(u >> 16);
  return __builtin_bit_cast(__bf16, s);
}
__device__ __forceinline__ float bf16_to_f(__bf16 h) {
  unsigned short s = __builtin_bit_cast(unsigned short, h);
  return __uint_as_float(((unsigned)s) << 16);
}
__device__ __forceinline__ v8f vzero8() {
  v8f a;
#pragma unroll
  for (int i = 0; i < 8; ++i) a[i] = 0.f;
  return a;
}
__device__ __forceinline__ v16bf load16bf(const __bf16* __restrict__ p) {
  v8bf a = *(const v8bf*)p;
  v8bf b = *(const v8bf*)(p + 8);
  return __builtin_shufflevector(a, b, 0, 1, 2, 3, 4, 5, 6, 7, 8, 9, 10, 11, 12,
                                 13, 14, 15);
}

// ---------------- WMMA: D = A*B + C, bf16 inputs, f32 accum ------------------
__device__ __forceinline__ v8f wmma_bf16(v16bf a, v16bf b, v8f c) {
  return __builtin_amdgcn_wmma_f32_16x16x32_bf16(false, a, false, b, (short)0, c,
                                                 false, false);
}
// bf16x3 emulation of fp32 GEMM: lo*hi + hi*lo + hi*hi (f32 accumulate)
__device__ __forceinline__ v8f mma3(v16bf ahi, v16bf alo, v16bf bhi, v16bf blo,
                                    v8f acc) {
  acc = wmma_bf16(alo, bhi, acc);
  acc = wmma_bf16(ahi, blo, acc);
  acc = wmma_bf16(ahi, bhi, acc);
  return acc;
}

// A fragment (16xK tile in LDS, row-major; contiguous k = k0 + 16*(lane/16) + e, same as B)
__device__ __forceinline__ void frag_a_f32(const float* __restrict__ A, int ldk,
                                           int k0, v16bf& hi, v16bf& lo) {
  const int lane = threadIdx.x & 31;
  const int m = lane & 15;
  const int kb = k0 + ((lane >> 4) << 4);
#pragma unroll
  for (int e = 0; e < 16; ++e) {
    int kk = kb + e;
    float f = A[m * ldk + kk];
    __bf16 h = to_bf16(f);
    hi[e] = h;
    lo[e] = to_bf16(f - bf16_to_f(h));
  }
}
// B fragment from TRANSPOSED pre-split weights Wt[n*K + k] (vector loads)
__device__ __forceinline__ void frag_b_wt(const __bf16* __restrict__ Whi_t,
                                          const __bf16* __restrict__ Wlo_t,
                                          int Krun, int k0, int n0, v16bf& hi,
                                          v16bf& lo) {
  const int lane = threadIdx.x & 31;
  const int n = n0 + (lane & 15);
  const int kb = k0 + ((lane >> 4) << 4);
  const size_t base = (size_t)n * Krun + kb;
  hi = load16bf(Whi_t + base);
  lo = load16bf(Wlo_t + base);
}

// ---------------- LDS staging (async global->LDS on CDNA5) -------------------
__device__ __forceinline__ void stage_lds_rows(const float* __restrict__ g,
                                               int gstride, float* l,
                                               int rowlen4, int rows, int tid,
                                               int nt) {
#if USE_ASYNC_LDS
  for (int c = tid; c < rows * rowlen4; c += nt) {
    int r = c / rowlen4, cc = (c % rowlen4) << 2;
    GASYNC_B128(g + (size_t)r * gstride + cc, l + r * (rowlen4 << 2) + cc);
  }
  __builtin_amdgcn_s_wait_asynccnt(0);
#else
  for (int c = tid; c < rows * rowlen4; c += nt) {
    int r = c / rowlen4, cc = (c % rowlen4) << 2;
    *(float4*)(l + r * (rowlen4 << 2) + cc) =
        *(const float4*)(g + (size_t)r * gstride + cc);
  }
#endif
}

// ---------------- small utility kernels --------------------------------------
__global__ void zero_kernel(float* p, long long n) {
  long long i = (long long)blockIdx.x * blockDim.x + threadIdx.x;
  if (i < n) p[i] = 0.f;
}
// split f32 weight [K][N] row-major into TRANSPOSED bf16 hi/lo [N][K]
__global__ void split_t_kernel(const float* __restrict__ src, __bf16* hi_t,
                               __bf16* lo_t, int K, int N) {
  int i = blockIdx.x * blockDim.x + threadIdx.x;
  if (i >= K * N) return;
  int k = i / N, n = i % N;
  float f = src[i];
  __bf16 h = to_bf16(f);
  size_t o = (size_t)n * K + k;
  hi_t[o] = h;
  lo_t[o] = to_bf16(f - bf16_to_f(h));
}

// ---------------- GEMM: C[M,Nc] = A[M,K]@W + bias ----------------------------
template <int K>
__global__ __launch_bounds__(256) void gemm_bias_kernel(
    const float* __restrict__ A, const __bf16* __restrict__ Whi_t,
    const __bf16* __restrict__ Wlo_t, const float* __restrict__ bias,
    float* __restrict__ C, int Nc) {
  extern __shared__ float As[];  // 16 x K (tile is contiguous: row stride == K)
  const int row0 = blockIdx.x * 16;
  const int tid = threadIdx.x;
  stage_lds_rows(A + (size_t)row0 * K, K, As, K / 4, 16, tid, blockDim.x);
  __syncthreads();
  const int wave = tid >> 5, lane = tid & 31;
  const int colsPerBlock = (blockDim.x >> 5) << 4;
  const int n0 = blockIdx.y * colsPerBlock + wave * 16;
  v8f acc = vzero8();
#pragma unroll
  for (int k0 = 0; k0 < K; k0 += 32) {
    v16bf ahi, alo, bhi, blo;
    frag_a_f32(As, K, k0, ahi, alo);
    frag_b_wt(Whi_t, Wlo_t, K, k0, n0, bhi, blo);
    acc = mma3(ahi, alo, bhi, blo, acc);
  }
  const int colg = n0 + (lane & 15);
  const int rb = (lane >> 4) << 3;
  const float bv = bias[colg];
#pragma unroll
  for (int r = 0; r < 8; ++r)
    C[(size_t)(row0 + rb + r) * Nc + colg] = acc[r] + bv;
}

// ---------------- input layer: relu(LN(x@W + b)) -----------------------------
template <int K>
__global__ __launch_bounds__(256) void gemm_ln_relu_kernel(
    const float* __restrict__ A, const __bf16* __restrict__ Whi_t,
    const __bf16* __restrict__ Wlo_t, const float* __restrict__ bias,
    const float* __restrict__ g, const float* __restrict__ bb,
    float* __restrict__ C) {
  extern __shared__ float sm[];
  float* As = sm;             // 16*K
  float* Cs = sm + 16 * K;    // 16*128
  float* mu = Cs + 16 * 128;  // 16
  float* iv = mu + 16;        // 16
  const int row0 = blockIdx.x * 16;
  const int tid = threadIdx.x;
  stage_lds_rows(A + (size_t)row0 * K, K, As, K / 4, 16, tid, 256);
  __syncthreads();
  const int wave = tid >> 5, lane = tid & 31;
  const int n0 = wave * 16;  // Nc == 128, 8 waves
  v8f acc = vzero8();
#pragma unroll
  for (int k0 = 0; k0 < K; k0 += 32) {
    v16bf ahi, alo, bhi, blo;
    frag_a_f32(As, K, k0, ahi, alo);
    frag_b_wt(Whi_t, Wlo_t, K, k0, n0, bhi, blo);
    acc = mma3(ahi, alo, bhi, blo, acc);
  }
  const int colg = n0 + (lane & 15);
  const int rb = (lane >> 4) << 3;
  const float bv = bias[colg];
#pragma unroll
  for (int r = 0; r < 8; ++r) Cs[(rb + r) * HID + colg] = acc[r] + bv;
  __syncthreads();
  if (tid < 16) {
    float s = 0.f, s2 = 0.f;
    for (int c = 0; c < HID; ++c) {
      float v = Cs[tid * HID + c];
      s += v;
      s2 += v * v;
    }
    float m = s * (1.f / HID);
    float var = s2 * (1.f / HID) - m * m;
    mu[tid] = m;
    iv[tid] = rsqrtf(var + LN_EPS);
  }
  __syncthreads();
  for (int i = tid; i < 16 * HID; i += 256) {
    int r = i >> 7, c = i & 127;
    float v = (Cs[i] - mu[r]) * iv[r] * g[c] + bb[c];
    C[(size_t)(row0 + r) * HID + c] = fmaxf(v, 0.f);
  }
}

// ---------------- global reductions over q,k,v -------------------------------
__global__ __launch_bounds__(256) void reduce_qkv_kernel(
    const float* __restrict__ q, const float* __restrict__ k,
    const float* __restrict__ v, float* scal, float* ksum, float* vsum,
    int rowsPerBlock) {
  __shared__ float red[256];
  const int c = threadIdx.x;
  const int n0 = blockIdx.x * rowsPerBlock;
  float ks = 0.f, vs = 0.f, sq = 0.f, sk = 0.f;
  for (int r = 0; r < rowsPerBlock; ++r) {
    int n = n0 + r;
    if (n >= NN) break;
    size_t idx = (size_t)n * QKV + c;
    float qv = q[idx], kv = k[idx], vv = v[idx];
    ks += kv;
    vs += vv;
    sq += qv * qv;
    sk += kv * kv;
  }
  atomicAdd(&ksum[c], ks);
  atomicAdd(&vsum[c], vs);
  red[c] = sq;
  __syncthreads();
  for (int o = 128; o; o >>= 1) {
    if (c < o) red[c] += red[c + o];
    __syncthreads();
  }
  if (c == 0) atomicAdd(&scal[0], red[0]);
  __syncthreads();
  red[c] = sk;
  __syncthreads();
  for (int o = 128; o; o >>= 1) {
    if (c < o) red[c] += red[c + o];
    __syncthreads();
  }
  if (c == 0) atomicAdd(&scal[1], red[0]);
}

// --------- kvs_t[h,d,m] = sum_n k[n,h,m]*v[n,h,d]  (split-K WMMA, transposed store)
#define KVS_NCH 32
#define KVS_CHUNK 1568  // 32*49; 32*KVS_CHUNK >= NN; only last chunk has a tail
template <bool GUARD>
__device__ __forceinline__ v8f kvs_accum(const float* __restrict__ k,
                                         const float* __restrict__ v, int nbase,
                                         int h, int m, int d, int kbA, int kbB,
                                         v8f acc) {
  for (int kk = 0; kk < KVS_CHUNK; kk += 32) {
    v16bf ahi, alo, bhi, blo;
#pragma unroll
    for (int e = 0; e < 16; ++e) {
      int n = nbase + kk + kbB + e;
      float f = (!GUARD || n < NN) ? k[(size_t)n * QKV + h * HID + m] : 0.f;
      __bf16 hh = to_bf16(f);
      ahi[e] = hh;
      alo[e] = to_bf16(f - bf16_to_f(hh));
      int n2 = nbase + kk + kbB + e;
      float gg = (!GUARD || n2 < NN) ? v[(size_t)n2 * QKV + h * HID + d] : 0.f;
      __bf16 gh = to_bf16(gg);
      bhi[e] = gh;
      blo[e] = to_bf16(gg - bf16_to_f(gh));
    }
    acc = mma3(ahi, alo, bhi, blo, acc);
  }
  return acc;
}
__global__ __launch_bounds__(32) void kvs_kernel(const float* __restrict__ k,
                                                 const float* __restrict__ v,
                                                 float* __restrict__ kvs_t) {
  const int m0 = blockIdx.x * 16, d0 = blockIdx.y * 16;
  const int h = blockIdx.z / KVS_NCH, ch = blockIdx.z % KVS_NCH;
  const int nbase = ch * KVS_CHUNK;
  const int lane = threadIdx.x & 31;
  const int m = m0 + (lane & 15);
  const int d = d0 + (lane & 15);
  const int kbA = (lane >> 4) << 3;
  const int kbB = (lane >> 4) << 4;
  v8f acc = vzero8();
  if (nbase + KVS_CHUNK <= NN)  // fast path: straight-line gather, no guards
    acc = kvs_accum<false>(k, v, nbase, h, m, d, kbA, kbB, acc);
  else  // tail chunk only
    acc = kvs_accum<true>(k, v, nbase, h, m, d, kbA, kbB, acc);
  const int rb = (lane >> 4) << 3;
#pragma unroll
  for (int r = 0; r < 8; ++r)  // transposed: [h][d][m]
    atomicAdd(&kvs_t[h * HID * HID + (size_t)d * HID + (m0 + rb + r)], acc[r]);
}

// ---------------- denom[n,h] = q.ksum/(|q||k|) + N ---------------------------
__global__ void denom_kernel(const float* __restrict__ q,
                             const float* __restrict__ ksum,
                             const float* __restrict__ scal,
                             float* __restrict__ denom) {
  int idx = blockIdx.x * blockDim.x + threadIdx.x;
  if (idx >= NN * HEADS) return;
  int n = idx >> 1, h = idx & 1;
  float s = 0.f;
  for (int j = 0; j < HID; ++j)
    s += q[(size_t)n * QKV + h * HID + j] * ksum[h * HID + j];
  float invqk = rsqrtf(scal[0] * scal[1]);
  denom[idx] = s * invqk + (float)NN;
}

// ---------------- attention: ag = (q@kvs * invqk + vsum)/denom ---------------
__global__ __launch_bounds__(256) void attn_kernel(
    const float* __restrict__ q, const float* __restrict__ kvs_t,
    const float* __restrict__ vsum, const float* __restrict__ denom,
    const float* __restrict__ scal, float* __restrict__ ag) {
  extern __shared__ float Aq[];  // 16 x 128
  const int row0 = blockIdx.x * 16;
  const int h = blockIdx.y;
  const int tid = threadIdx.x;
  stage_lds_rows(q + (size_t)row0 * QKV + h * HID, QKV, Aq, HID / 4, 16, tid,
                 256);
  __syncthreads();
  const int wave = tid >> 5, lane = tid & 31;
  const int d0 = wave * 16;
  v8f acc = vzero8();
#pragma unroll
  for (int k0 = 0; k0 < HID; k0 += 32) {
    v16bf ahi, alo, bhi, blo;
    frag_a_f32(Aq, HID, k0, ahi, alo);
    {  // B[k][d] = kvs_t[h][d][k]: 16 contiguous floats per fragment
      int d = d0 + (lane & 15);
      int kb = k0 + ((lane >> 4) << 4);
      const float* p = kvs_t + h * HID * HID + (size_t)d * HID + kb;
#pragma unroll
      for (int e = 0; e < 16; ++e) {
        float f = p[e];
        __bf16 hh = to_bf16(f);
        bhi[e] = hh;
        blo[e] = to_bf16(f - bf16_to_f(hh));
      }
    }
    acc = mma3(ahi, alo, bhi, blo, acc);
  }
  const float invqk = rsqrtf(scal[0] * scal[1]);
  const int col = d0 + (lane & 15);
  const int rb = (lane >> 4) << 3;
#pragma unroll
  for (int r = 0; r < 8; ++r) {
    int n = row0 + rb + r;
    float val = (acc[r] * invqk + vsum[h * HID + col]) / denom[n * 2 + h];
    ag[(size_t)n * QKV + h * HID + col] = val;
  }
}

// ---------------- graph: degree + symmetric-normalized scatter ---------------
__global__ void deg_kernel(const int* __restrict__ ecol, float* __restrict__ deg) {
  int e = blockIdx.x * blockDim.x + threadIdx.x;
  if (e < EE) atomicAdd(&deg[ecol[e]], 1.f);
}
__global__ void gcn_kernel(const float* __restrict__ v,
                           const int* __restrict__ erow,
                           const int* __restrict__ ecol,
                           const float* __restrict__ deg,
                           float* __restrict__ ag) {
  long long t = (long long)blockIdx.x * blockDim.x + threadIdx.x;
  if (t >= (long long)EE * 64) return;
  int e = (int)(t >> 6);
  int j = ((int)t & 63) << 2;
  int r = erow[e], c = ecol[e];
  float dr = deg[r], dc = deg[c];
  float w = (dr > 0.f && dc > 0.f) ? rsqrtf(dr) * rsqrtf(dc) : 0.f;
  const float4 vv = *(const float4*)(v + (size_t)r * QKV + j);
  float* dst = ag + (size_t)c * QKV + j;
  atomicAdd(dst + 0, w * vv.x);
  atomicAdd(dst + 1, w * vv.y);
  atomicAdd(dst + 2, w * vv.z);
  atomicAdd(dst + 3, w * vv.w);
}

// ---------------- adaptive eta pieces ----------------------------------------
__global__ __launch_bounds__(256) void ekin_kernel(const float* __restrict__ h,
                                                   const float* __restrict__ prev,
                                                   float* scal) {
  __shared__ float red[256];
  long long i = (long long)blockIdx.x * 256 + threadIdx.x;
  float s = 0.f;
  if (i < (long long)NN * HID) {
    float d = h[i] - prev[i];
    s = d * d;
  }
  red[threadIdx.x] = s;
  __syncthreads();
  for (int o = 128; o; o >>= 1) {
    if (threadIdx.x < o) red[threadIdx.x] += red[threadIdx.x + o];
    __syncthreads();
  }
  if (threadIdx.x == 0) atomicAdd(&scal[2], 0.5f * red[0]);
}
__global__ __launch_bounds__(256) void epot_kernel(const float* __restrict__ h,
                                                   const int* __restrict__ erow,
                                                   const int* __restrict__ ecol,
                                                   float* scal) {
  int wid = (blockIdx.x << 3) + (threadIdx.x >> 5);  // one edge per wave
  int lane = threadIdx.x & 31;
  if (wid >= EE) return;
  int r = erow[wid], c = ecol[wid];
  const float4 a = *(const float4*)(h + (size_t)r * HID + (lane << 2));
  const float4 b = *(const float4*)(h + (size_t)c * HID + (lane << 2));
  float dx = a.x - b.x, dy = a.y - b.y, dz = a.z - b.z, dw = a.w - b.w;
  float s = dx * dx + dy * dy + dz * dz + dw * dw;
  for (int o = 16; o; o >>= 1) s += __shfl_xor(s, o, 32);
  if (lane == 0) atomicAdd(&scal[3], 0.5f * (float)(r + c) * s);
}
__global__ void eta_kernel(float* scal, const float* __restrict__ A1,
                           const float* __restrict__ b1,
                           const float* __restrict__ A2,
                           const float* __restrict__ b2) {
  if (threadIdx.x != 0 || blockIdx.x != 0) return;
  float r = scal[2] / (scal[3] + 1e-8f);
  float dr = r - 1.f;
  float acc = 0.f;
  for (int j = 0; j < 16; ++j) {
    float hj = fmaxf(dr * A1[j] + b1[j], 0.f);
    acc += hj * A2[j];
  }
  scal[4] = 1.f / (1.f + expf(-(acc + b2[0])));
}

// ---------------- combine: new_h = LN(0.5*(Lhl + eta*(h-prev)) + 0.5*h) ------
__global__ __launch_bounds__(128) void combine_kernel(
    const float* __restrict__ ag, const float* __restrict__ h,
    const float* __restrict__ prev, const float* __restrict__ scal,
    const float* __restrict__ g, const float* __restrict__ bb,
    float* __restrict__ outp) {
  __shared__ float s1[128], s2[128];
  const int n = blockIdx.x, d = threadIdx.x;
  const float eta = scal[4];
  const float hv = h[(size_t)n * HID + d];
  const float lhl =
      0.5f * (ag[(size_t)n * QKV + d] + ag[(size_t)n * QKV + HID + d]);
  const float val =
      0.5f * (lhl + eta * (hv - prev[(size_t)n * HID + d])) + 0.5f * hv;
  s1[d] = val;
  s2[d] = val * val;
  __syncthreads();
  for (int o = 64; o; o >>= 1) {
    if (d < o) {
      s1[d] += s1[d + o];
      s2[d] += s2[d + o];
    }
    __syncthreads();
  }
  float m = s1[0] * (1.f / HID);
  float var = s2[0] * (1.f / HID) - m * m;
  outp[(size_t)n * HID + d] = g[d] * (val - m) * rsqrtf(var + LN_EPS) + bb[d];
}

// ---------------- host orchestration -----------------------------------------
static inline int cdiv_ll(long long a, long long b) { return (int)((a + b - 1) / b); }

extern "C" void kernel_launch(void* const* d_in, const int* in_sizes, int n_in,
                              void* d_out, int out_size, void* d_ws,
                              size_t ws_size, hipStream_t stream) {
  (void)in_sizes; (void)n_in; (void)out_size; (void)ws_size;
  const float* x    = (const float*)d_in[0];
  const int*   edge = (const int*)d_in[1];  // [2,E] int32
  const float* in_w = (const float*)d_in[2];
  const float* in_b = (const float*)d_in[3];
  const float* Wq_w = (const float*)d_in[4];
  const float* Wq_b = (const float*)d_in[5];
  const float* Wk_w = (const float*)d_in[6];
  const float* Wk_b = (const float*)d_in[7];
  const float* Wv_w = (const float*)d_in[8];
  const float* Wv_b = (const float*)d_in[9];
  const float* ln_w = (const float*)d_in[10];
  const float* ln_b = (const float*)d_in[11];
  const float* out_w = (const float*)d_in[12];
  const float* out_b = (const float*)d_in[13];
  const float* A1 = (const float*)d_in[14];
  const float* b1 = (const float*)d_in[15];
  const float* A2 = (const float*)d_in[16];
  const float* b2 = (const float*)d_in[17];
  float* Y = (float*)d_out;
  const int* erow = edge;
  const int* ecol = edge + EE;

  char* base = (char*)d_ws;
  size_t off = 0;
  auto take = [&](size_t bytes) -> void* {
    void* p = base + off;
    off += bytes;
    off = (off + 255) & ~(size_t)255;
    return p;
  };
  float* buf0 = (float*)take((size_t)NN * HID * 4);
  float* buf1 = (float*)take((size_t)NN * HID * 4);
  float* buf2 = (float*)take((size_t)NN * HID * 4);
  float* qb = (float*)take((size_t)NN * QKV * 4);
  float* kb = (float*)take((size_t)NN * QKV * 4);
  float* vb = (float*)take((size_t)NN * QKV * 4);
  float* ag = (float*)take((size_t)NN * QKV * 4);
  float* deg = (float*)take((size_t)NN * 4);
  float* denom = (float*)take((size_t)NN * HEADS * 4);
  float* kvs = (float*)take((size_t)HEADS * HID * HID * 4);
  float* ksum = (float*)take((size_t)HEADS * HID * 4);
  float* vsum = (float*)take((size_t)HEADS * HID * 4);
  float* scal = (float*)take(64 * 4);
  __bf16* inw_hi = (__bf16*)take((size_t)INF_ * HID * 2);
  __bf16* inw_lo = (__bf16*)take((size_t)INF_ * HID * 2);
  __bf16* qkv_hi = (__bf16*)take((size_t)6 * HID * QKV * 2);
  __bf16* qkv_lo = (__bf16*)take((size_t)6 * HID * QKV * 2);
  __bf16* outw_hi = (__bf16*)take((size_t)HID * OUTF * 2);
  __bf16* outw_lo = (__bf16*)take((size_t)HID * OUTF * 2);

  // 1) weight prep: split into bf16 hi/lo, stored transposed [N][K]
  split_t_kernel<<<cdiv_ll(INF_ * HID, 256), 256, 0, stream>>>(
      in_w, inw_hi, inw_lo, INF_, HID);
  for (int i = 0; i < 2; ++i) {
    split_t_kernel<<<cdiv_ll(HID * QKV, 256), 256, 0, stream>>>(
        Wq_w + (size_t)i * HID * QKV, qkv_hi + (size_t)(0 * 2 + i) * HID * QKV,
        qkv_lo + (size_t)(0 * 2 + i) * HID * QKV, HID, QKV);
    split_t_kernel<<<cdiv_ll(HID * QKV, 256), 256, 0, stream>>>(
        Wk_w + (size_t)i * HID * QKV, qkv_hi + (size_t)(1 * 2 + i) * HID * QKV,
        qkv_lo + (size_t)(1 * 2 + i) * HID * QKV, HID, QKV);
    split_t_kernel<<<cdiv_ll(HID * QKV, 256), 256, 0, stream>>>(
        Wv_w + (size_t)i * HID * QKV, qkv_hi + (size_t)(2 * 2 + i) * HID * QKV,
        qkv_lo + (size_t)(2 * 2 + i) * HID * QKV, HID, QKV);
  }
  split_t_kernel<<<cdiv_ll(HID * OUTF, 256), 256, 0, stream>>>(
      out_w, outw_hi, outw_lo, HID, OUTF);

  // 2) init
  zero_kernel<<<cdiv_ll((long long)NN * HID, 256), 256, 0, stream>>>(
      buf0, (long long)NN * HID);  // prev = 0
  zero_kernel<<<cdiv_ll(NN, 256), 256, 0, stream>>>(deg, NN);
  deg_kernel<<<cdiv_ll(EE, 256), 256, 0, stream>>>(ecol, deg);

  // 3) input layer: h = relu(LN(x@in_w + in_b))
  gemm_ln_relu_kernel<INF_>
      <<<dim3(NN / 16), 256, (16 * INF_ + 16 * HID + 32) * sizeof(float),
         stream>>>(x, inw_hi, inw_lo, in_b, ln_w, ln_b, buf1);

  float* prevP = buf0;
  float* hP = buf1;
  float* spare = buf2;
  for (int i = 0; i < 2; ++i) {
    zero_kernel<<<1, 64, 0, stream>>>(scal, 16);
    zero_kernel<<<1, 256, 0, stream>>>(ksum, HEADS * HID);
    zero_kernel<<<1, 256, 0, stream>>>(vsum, HEADS * HID);
    zero_kernel<<<cdiv_ll(HEADS * HID * HID, 256), 256, 0, stream>>>(
        kvs, HEADS * HID * HID);

    const size_t wsh = 16 * HID * sizeof(float);
    gemm_bias_kernel<HID><<<dim3(NN / 16, 2), 256, wsh, stream>>>(
        hP, qkv_hi + (size_t)(0 * 2 + i) * HID * QKV,
        qkv_lo + (size_t)(0 * 2 + i) * HID * QKV, Wq_b + (size_t)i * QKV, qb,
        QKV);
    gemm_bias_kernel<HID><<<dim3(NN / 16, 2), 256, wsh, stream>>>(
        hP, qkv_hi + (size_t)(1 * 2 + i) * HID * QKV,
        qkv_lo + (size_t)(1 * 2 + i) * HID * QKV, Wk_b + (size_t)i * QKV, kb,
        QKV);
    gemm_bias_kernel<HID><<<dim3(NN / 16, 2), 256, wsh, stream>>>(
        hP, qkv_hi + (size_t)(2 * 2 + i) * HID * QKV,
        qkv_lo + (size_t)(2 * 2 + i) * HID * QKV, Wv_b + (size_t)i * QKV, vb,
        QKV);

    reduce_qkv_kernel<<<200, 256, 0, stream>>>(qb, kb, vb, scal, ksum, vsum, 250);
    kvs_kernel<<<dim3(8, 8, HEADS * KVS_NCH), 32, 0, stream>>>(kb, vb, kvs);
    denom_kernel<<<cdiv_ll(NN * HEADS, 256), 256, 0, stream>>>(qb, ksum, scal,
                                                               denom);
    attn_kernel<<<dim3(NN / 16, HEADS), 256, 16 * HID * sizeof(float), stream>>>(
        qb, kvs, vsum, denom, scal, ag);
    gcn_kernel<<<cdiv_ll((long long)EE * 64, 256), 256, 0, stream>>>(
        vb, erow, ecol, deg, ag);
    ekin_kernel<<<cdiv_ll((long long)NN * HID, 256), 256, 0, stream>>>(hP, prevP,
                                                                       scal);
    epot_kernel<<<cdiv_ll(EE, 8), 256, 0, stream>>>(hP, erow, ecol, scal);
    eta_kernel<<<1, 32, 0, stream>>>(scal, A1, b1, A2, b2);
    combine_kernel<<<NN, 128, 0, stream>>>(ag, hP, prevP, scal,
                                           ln_w + (size_t)(i + 1) * HID,
                                           ln_b + (size_t)(i + 1) * HID, spare);
    float* t = prevP;
    prevP = hP;
    hP = spare;
    spare = t;
  }

  // 4) output projection: Y = h @ out_w + out_b  (Nc=64, 4 waves)
  gemm_bias_kernel<HID><<<dim3(NN / 16, 1), 128, 16 * HID * sizeof(float),
                          stream>>>(hP, outw_hi, outw_lo, out_b, Y, OUTF);
}
